// ResidualSwinBlock_68315749810365
// MI455X (gfx1250) — matrix operand order repaired and checked by reading, not compiled, not run on hardware
//
#include <hip/hip_runtime.h>
#include <hip/hip_bf16.h>
#include <math.h>

typedef __attribute__((ext_vector_type(16))) _Float16 v16h;
typedef __attribute__((ext_vector_type(8)))  _Float16 v8h;
typedef __attribute__((ext_vector_type(8)))  float    v8f;

#define BQ     8
#define CDIM   192
#define HH     128
#define WW_    128
#define NH     6
#define HD     32
#define NTOK   64
#define HID    768
#define SHIFT_ 4

// ---- LDS strides (elements) ----
#define SXB 193   // f32 token-major arrays (bank-spread: 193 mod 64 = 1)
#define SA  200   // f16 token-major arrays (mult of 8 halves -> 16B rows)
#define SKT 72    // kT rows (mult of 8, rows 144B -> 16B aligned)
#define SS  72    // per-head S/P rows
#define SM  776   // fc1 activation rows (mult of 8, 1552B rows)

// ---- LDS offsets (bytes), phase-aliased ----
#define OFF_XB32 0                 // 64*193*4 = 49408   (persistent shortcut/residual)
#define OFF_R1   49408             // 25600: xs16 -> attnO16 -> t16
#define OFF_Q    75008             // 25600: q16; later t32 (49408) spans Q+KT
#define OFF_KT   100608            // 27648: kT16
#define OFF_T32  75008
#define OFF_V    128256            // 25600: v16; later m16 (99328) spans V+S
#define OFF_S    153856            // 55296: S16/P16
#define OFF_M16  128256
#define SMEM_BYTES 227584

// ---- workspace offsets (bytes) ----
#define WS_STATS 0                 // 16 floats: mean[8], rstd[8]
#define WS_BIAS  256               // 6*64*64 f32 = 98304
#define WS_QKV   98816             // 110592 halves
#define WS_PROJ  320000            // 36864 halves
#define WS_FC1   393728            // 147456 halves
#define WS_FC2   688640            // 147456 halves
#define WS_PART  983552            // 512*2 doubles

__device__ __forceinline__ v8f wmma32(v16h a, v16h b, v8f c) {
  return __builtin_amdgcn_wmma_f32_16x16x32_f16(false, a, false, b, (short)0, c,
                                                false, false);
}

// A fragment 16x32 f16 from row-major [M][stride] (LDS): lane m=l&15, g=l>>4,
// halves 0..7 = K k0+8g.., halves 8..15 = K k0+16+8g..   (ISA 7.12.2)
__device__ __forceinline__ v16h ldsA(const _Float16* base, int m0, int k0, int stride) {
  int lane = threadIdx.x & 31;
  int g = lane >> 4, m = (lane & 15) + m0;
  const _Float16* p = base + m * stride + k0 + 8 * g;
  union { v16h v; v8h h[2]; } u;
  u.h[0] = *(const v8h*)p;
  u.h[1] = *(const v8h*)(p + 16);
  return u.v;
}

// B fragment 32x16 f16 from row-major [K][stride]: lane n=l&15, g=l>>4,
// halves 0..7 = K k0+8g.., halves 8..15 = K k0+16+8g.. (same as A).
__device__ __forceinline__ v16h ldB(const _Float16* base, int k0, int n0, int stride) {
  int lane = threadIdx.x & 31;
  int g = lane >> 4, n = (lane & 15) + n0;
  const _Float16* p = base + (k0 + 8 * g) * stride + n;
  v16h v;
  for (int s = 0; s < 8; ++s) {
    v[s] = p[s * stride];
    v[8 + s] = p[(16 + s) * stride];
  }
  return v;
}

__device__ __forceinline__ int regid(int y) { return y < 120 ? 0 : (y < 124 ? 1 : 2); }

// ---------------- GroupNorm stats (two stage) ----------------
__global__ void gn_part_kernel(const float* __restrict__ x, double* __restrict__ part) {
  int b = blockIdx.x >> 6, sl = blockIdx.x & 63;
  const float* p = x + (size_t)b * (CDIM * HH * WW_) + (size_t)sl * 49152;
  double s = 0.0, s2 = 0.0;
  for (int i = threadIdx.x; i < 49152; i += 256) {
    float v = p[i]; s += v; s2 += (double)v * (double)v;
  }
  __shared__ double sh[512];
  sh[threadIdx.x] = s; sh[256 + threadIdx.x] = s2;
  __syncthreads();
  for (int o = 128; o > 0; o >>= 1) {
    if ((int)threadIdx.x < o) {
      sh[threadIdx.x] += sh[threadIdx.x + o];
      sh[256 + threadIdx.x] += sh[256 + threadIdx.x + o];
    }
    __syncthreads();
  }
  if (threadIdx.x == 0) { part[2 * blockIdx.x] = sh[0]; part[2 * blockIdx.x + 1] = sh[256]; }
}

__global__ void gn_fin_kernel(const double* __restrict__ part, float* __restrict__ stats) {
  int b = blockIdx.x;
  __shared__ double sh[128];
  sh[threadIdx.x]      = part[2 * (b * 64 + threadIdx.x)];
  sh[64 + threadIdx.x] = part[2 * (b * 64 + threadIdx.x) + 1];
  __syncthreads();
  for (int o = 32; o > 0; o >>= 1) {
    if ((int)threadIdx.x < o) {
      sh[threadIdx.x] += sh[threadIdx.x + o];
      sh[64 + threadIdx.x] += sh[64 + threadIdx.x + o];
    }
    __syncthreads();
  }
  if (threadIdx.x == 0) {
    double n = (double)CDIM * HH * WW_;
    double mu = sh[0] / n;
    double var = sh[64] / n - mu * mu;
    stats[b] = (float)mu;
    stats[8 + b] = (float)(1.0 / sqrt(var + 1e-5));
  }
}

// ---------------- rel-pos bias table: [NH][64][64] ----------------
__global__ void bias_kernel(const float* __restrict__ rpb, float* __restrict__ tab) {
  int idx = blockIdx.x * blockDim.x + threadIdx.x;
  if (idx >= NH * NTOK * NTOK) return;
  int h = idx >> 12, r = idx & 4095, i = r >> 6, j = r & 63;
  int dy = (i >> 3) - (j >> 3) + 7;
  int dx = (i & 7) - (j & 7) + 7;
  tab[idx] = rpb[(dy * 15 + dx) * NH + h];
}

// ---------------- f32 -> f16 weight convert ----------------
__global__ void cvt_kernel(const float* __restrict__ s, _Float16* __restrict__ d, int n) {
  int i = blockIdx.x * blockDim.x + threadIdx.x;
  if (i < n) d[i] = (_Float16)s[i];
}

// ---------------- fused Swin block: one workgroup per shifted window ----------------
__global__ void __launch_bounds__(256)
swin_kernel(const float* __restrict__ x,
            const float* __restrict__ gn_w,  const float* __restrict__ gn_b,
            const float* __restrict__ ln1_w, const float* __restrict__ ln1_b,
            const float* __restrict__ qkv_b, const float* __restrict__ proj_b,
            const float* __restrict__ ln2_w, const float* __restrict__ ln2_b,
            const float* __restrict__ fc1_b, const float* __restrict__ fc2_b,
            const _Float16* __restrict__ qkv16, const _Float16* __restrict__ proj16,
            const _Float16* __restrict__ fc116, const _Float16* __restrict__ fc216,
            const float* __restrict__ biasTab, const float* __restrict__ stats,
            float* __restrict__ out) {
  extern __shared__ char smem[];
  float*    xb32 = (float*)(smem + OFF_XB32);
  _Float16* xs16 = (_Float16*)(smem + OFF_R1);
  _Float16* q16  = (_Float16*)(smem + OFF_Q);
  _Float16* kT16 = (_Float16*)(smem + OFF_KT);
  _Float16* v16_ = (_Float16*)(smem + OFF_V);
  _Float16* s16  = (_Float16*)(smem + OFF_S);
  _Float16* ao16 = (_Float16*)(smem + OFF_R1);   // aliases xs16 (dead)
  float*    t32  = (float*)(smem + OFF_T32);     // aliases q16+kT16 (dead)
  _Float16* t16  = (_Float16*)(smem + OFF_R1);   // aliases attnO (dead)
  _Float16* m16  = (_Float16*)(smem + OFF_M16);  // aliases v16+S16 (dead)

  const int tid  = threadIdx.x;
  const int lane = tid & 31;
  const int wv   = tid >> 5;
  const int gl   = lane >> 4;
  const int nn   = lane & 15;

  const int bid  = blockIdx.x;
  const int b    = bid >> 8;
  const int widx = bid & 255;
  const int wh   = widx >> 4;
  const int ww   = widx & 15;

  const float mean = stats[b];
  const float rstd = stats[8 + b];

  // ---- Phase 0: global load + shift-roll + GroupNorm -> xb32 (shortcut) ----
  for (int idx = tid; idx < NTOK * CDIM; idx += 256) {
    int tok = idx & 63, c = idx >> 6;
    int oy = (wh * 8 + (tok >> 3) + SHIFT_) & 127;
    int ox = (ww * 8 + (tok & 7) + SHIFT_) & 127;
    float v = x[(((size_t)b * CDIM + c) * HH + oy) * WW_ + ox];
    xb32[tok * SXB + c] = (v - mean) * rstd * gn_w[c] + gn_b[c];
  }
  __syncthreads();

  // ---- Phase 0b: LayerNorm1 -> xs16 ----
  if (tid < NTOK) {
    float s = 0.f, s2 = 0.f;
    for (int c = 0; c < CDIM; ++c) { float v = xb32[tid * SXB + c]; s += v; s2 += v * v; }
    float mu = s / CDIM;
    float rs = rsqrtf(s2 / CDIM - mu * mu + 1e-5f);
    for (int c = 0; c < CDIM; ++c) {
      float v = (xb32[tid * SXB + c] - mu) * rs * ln1_w[c] + ln1_b[c];
      xs16[tid * SA + c] = (_Float16)v;
    }
  }
  __syncthreads();

  // ---- Phase 1: QKV GEMM [64,192]x[192,576], K transposed on store ----
  // tn is wave-uniform -> scatter target (q / kT / v) resolved with scalar branches.
  for (int t = wv; t < 4 * 36; t += 8) {
    int tm = t / 36, tn = t % 36;
    v8f acc = {};
    for (int kt = 0; kt < 6; ++kt) {
      v16h a  = ldsA(xs16, tm * 16, kt * 32, SA);
      v16h bb = ldB(qkv16, kt * 32, tn * 16, 3 * CDIM);
      acc = wmma32(a, bb, acc);
    }
    int col = tn * 16 + nn;
    float bias = qkv_b[col];
    if (tn < 12) {                 // Q columns [0,192)
      for (int i = 0; i < 8; ++i) {
        int m = tm * 16 + i + 8 * gl;
        q16[m * SA + col] = (_Float16)(acc[i] + bias);
      }
    } else if (tn < 24) {          // K columns [192,384) -> transposed store
      int d = col - CDIM;
      for (int i = 0; i < 8; ++i) {
        int m = tm * 16 + i + 8 * gl;
        kT16[d * SKT + m] = (_Float16)(acc[i] + bias);
      }
    } else {                       // V columns [384,576)
      int d = col - 2 * CDIM;
      for (int i = 0; i < 8; ++i) {
        int m = tm * 16 + i + 8 * gl;
        v16_[m * SA + d] = (_Float16)(acc[i] + bias);
      }
    }
  }
  __syncthreads();

  // ---- Phase 2: S = Q K^T per head (K = hd = 32 -> 1 wmma/tile) ----
  for (int t = wv; t < NH * 16; t += 8) {
    int h = t >> 4, tt = t & 15, tm = tt >> 2, tn = tt & 3;
    v16h a  = ldsA(q16, tm * 16, h * HD, SA);
    v16h bb = ldB(kT16, h * HD, tn * 16, SKT);
    v8f acc = {};
    acc = wmma32(a, bb, acc);
    for (int i = 0; i < 8; ++i) {
      int m = tm * 16 + i + 8 * gl;
      s16[(h * NTOK + m) * SS + tn * 16 + nn] = (_Float16)acc[i];
    }
  }
  __syncthreads();

  // ---- Phase 3: softmax(S*scale + relpos bias + shift mask) in place ----
  {
    const float scl = 0.17677669529663687f;  // 1/sqrt(32)
    for (int r = tid; r < NH * NTOK; r += 256) {
      int h = r >> 6, m = r & 63;
      _Float16* row = s16 + (h * NTOK + m) * SS;
      const float* tb = biasTab + (h * NTOK + m) * NTOK;
      int idi = 3 * regid(wh * 8 + (m >> 3)) + regid(ww * 8 + (m & 7));
      float mx = -3.0e38f;
      for (int j = 0; j < NTOK; ++j) {
        int idj = 3 * regid(wh * 8 + (j >> 3)) + regid(ww * 8 + (j & 7));
        float l = (float)row[j] * scl + tb[j] + (idi == idj ? 0.f : -100.f);
        mx = fmaxf(mx, l);
      }
      float sum = 0.f;
      for (int j = 0; j < NTOK; ++j) {
        int idj = 3 * regid(wh * 8 + (j >> 3)) + regid(ww * 8 + (j & 7));
        float l = (float)row[j] * scl + tb[j] + (idi == idj ? 0.f : -100.f);
        float e = __expf(l - mx);
        sum += e;
        row[j] = (_Float16)e;
      }
      float inv = 1.f / sum;
      for (int j = 0; j < NTOK; ++j) row[j] = (_Float16)((float)row[j] * inv);
    }
  }
  __syncthreads();

  // ---- Phase 4: O = P V per head ----
  for (int t = wv; t < NH * 8; t += 8) {
    int h = t >> 3, tt = t & 7, tm = tt >> 1, tn = tt & 1;
    v8f acc = {};
    for (int kt = 0; kt < 2; ++kt) {
      v16h a  = ldsA(s16 + h * NTOK * SS, tm * 16, kt * 32, SS);
      v16h bb = ldB(v16_, kt * 32, h * HD + tn * 16, SA);
      acc = wmma32(a, bb, acc);
    }
    for (int i = 0; i < 8; ++i) {
      int m = tm * 16 + i + 8 * gl;
      ao16[m * SA + h * HD + tn * 16 + nn] = (_Float16)acc[i];
    }
  }
  __syncthreads();

  // ---- Phase 5: proj GEMM + shortcut -> t32 ----
  for (int t = wv; t < 4 * 12; t += 8) {
    int tm = t / 12, tn = t % 12;
    v8f acc = {};
    for (int kt = 0; kt < 6; ++kt) {
      v16h a  = ldsA(ao16, tm * 16, kt * 32, SA);
      v16h bb = ldB(proj16, kt * 32, tn * 16, CDIM);
      acc = wmma32(a, bb, acc);
    }
    int col = tn * 16 + nn;
    float bias = proj_b[col];
    for (int i = 0; i < 8; ++i) {
      int m = tm * 16 + i + 8 * gl;
      t32[m * SXB + col] = xb32[m * SXB + col] + acc[i] + bias;
    }
  }
  __syncthreads();

  // ---- Phase 5b: LayerNorm2 -> t16 ----
  if (tid < NTOK) {
    float s = 0.f, s2 = 0.f;
    for (int c = 0; c < CDIM; ++c) { float v = t32[tid * SXB + c]; s += v; s2 += v * v; }
    float mu = s / CDIM;
    float rs = rsqrtf(s2 / CDIM - mu * mu + 1e-5f);
    for (int c = 0; c < CDIM; ++c) {
      float v = (t32[tid * SXB + c] - mu) * rs * ln2_w[c] + ln2_b[c];
      t16[tid * SA + c] = (_Float16)v;
    }
  }
  __syncthreads();

  // ---- Phase 6: fc1 + exact GELU -> m16 ----
  for (int t = wv; t < 4 * 48; t += 8) {
    int tm = t / 48, tn = t % 48;
    v8f acc = {};
    for (int kt = 0; kt < 6; ++kt) {
      v16h a  = ldsA(t16, tm * 16, kt * 32, SA);
      v16h bb = ldB(fc116, kt * 32, tn * 16, HID);
      acc = wmma32(a, bb, acc);
    }
    int col = tn * 16 + nn;
    float bias = fc1_b[col];
    for (int i = 0; i < 8; ++i) {
      int m = tm * 16 + i + 8 * gl;
      float v = acc[i] + bias;
      v = 0.5f * v * (1.f + erff(v * 0.70710678118f));
      m16[m * SM + col] = (_Float16)v;
    }
  }
  __syncthreads();

  // ---- Phase 7: fc2 + residual + wrapper residual + clamp -> t32 in place ----
  for (int t = wv; t < 4 * 12; t += 8) {
    int tm = t / 12, tn = t % 12;
    v8f acc = {};
    for (int kt = 0; kt < 24; ++kt) {
      v16h a  = ldsA(m16, tm * 16, kt * 32, SM);
      v16h bb = ldB(fc216, kt * 32, tn * 16, CDIM);
      acc = wmma32(a, bb, acc);
    }
    int col = tn * 16 + nn;
    float bias = fc2_b[col];
    for (int i = 0; i < 8; ++i) {
      int m = tm * 16 + i + 8 * gl;
      float y = t32[m * SXB + col] + acc[i] + bias + xb32[m * SXB + col];
      t32[m * SXB + col] = fminf(fmaxf(y, -10.f), 10.f);
    }
  }
  __syncthreads();

  // ---- Phase 8: coalesced store with reverse shift ----
  for (int idx = tid; idx < NTOK * CDIM; idx += 256) {
    int tok = idx & 63, c = idx >> 6;
    int oy = (wh * 8 + (tok >> 3) + SHIFT_) & 127;
    int ox = (ww * 8 + (tok & 7) + SHIFT_) & 127;
    out[(((size_t)b * CDIM + c) * HH + oy) * WW_ + ox] = t32[tok * SXB + c];
  }
}

extern "C" void kernel_launch(void* const* d_in, const int* in_sizes, int n_in,
                              void* d_out, int out_size, void* d_ws, size_t ws_size,
                              hipStream_t stream) {
  (void)in_sizes; (void)n_in; (void)out_size; (void)ws_size;
  const float* x      = (const float*)d_in[0];
  const float* gn_w   = (const float*)d_in[1];
  const float* gn_b   = (const float*)d_in[2];
  const float* ln1_w  = (const float*)d_in[3];
  const float* ln1_b  = (const float*)d_in[4];
  const float* qkv_w  = (const float*)d_in[5];
  const float* qkv_b  = (const float*)d_in[6];
  const float* proj_w = (const float*)d_in[7];
  const float* proj_b = (const float*)d_in[8];
  const float* rpb    = (const float*)d_in[9];
  const float* ln2_w  = (const float*)d_in[10];
  const float* ln2_b  = (const float*)d_in[11];
  const float* fc1_w  = (const float*)d_in[12];
  const float* fc1_b  = (const float*)d_in[13];
  const float* fc2_w  = (const float*)d_in[14];
  const float* fc2_b  = (const float*)d_in[15];
  float* out = (float*)d_out;
  char*  ws  = (char*)d_ws;

  float*    stats  = (float*)(ws + WS_STATS);
  float*    btab   = (float*)(ws + WS_BIAS);
  _Float16* qkv16  = (_Float16*)(ws + WS_QKV);
  _Float16* proj16 = (_Float16*)(ws + WS_PROJ);
  _Float16* fc116  = (_Float16*)(ws + WS_FC1);
  _Float16* fc216  = (_Float16*)(ws + WS_FC2);
  double*   part   = (double*)(ws + WS_PART);

  gn_part_kernel<<<512, 256, 0, stream>>>(x, part);
  gn_fin_kernel<<<8, 64, 0, stream>>>(part, stats);
  bias_kernel<<<96, 256, 0, stream>>>(rpb, btab);
  cvt_kernel<<<432, 256, 0, stream>>>(qkv_w, qkv16, CDIM * 3 * CDIM);
  cvt_kernel<<<144, 256, 0, stream>>>(proj_w, proj16, CDIM * CDIM);
  cvt_kernel<<<576, 256, 0, stream>>>(fc1_w, fc116, CDIM * HID);
  cvt_kernel<<<576, 256, 0, stream>>>(fc2_w, fc216, HID * CDIM);

  (void)hipFuncSetAttribute(reinterpret_cast<const void*>(swin_kernel),
                            hipFuncAttributeMaxDynamicSharedMemorySize, SMEM_BYTES);
  swin_kernel<<<BQ * 16 * 16, 256, SMEM_BYTES, stream>>>(
      x, gn_w, gn_b, ln1_w, ln1_b, qkv_b, proj_b, ln2_w, ln2_b, fc1_b, fc2_b,
      qkv16, proj16, fc116, fc216, btab, stats, out);
}
